// StandardAttention_68341519614166
// MI455X (gfx1250) — matrix operand order repaired and checked by reading, not compiled, not run on hardware
//
#include <hip/hip_runtime.h>

// ---------------------------------------------------------------------------
// CDNA5 (gfx1250, wave32) fused multi-head attention
//   qkv = x @ w_qkv + b      (WMMA f16 -> f32, async-LDS double-buffered)
//   flash attention per head (WMMA f16 -> f32, online softmax, key mask)
//   out = attn @ w_out + b   (WMMA f16 -> f32, async-LDS double-buffered)
// Weights are pre-transposed once so every LDS staging copy is a contiguous
// 16-byte chunk -> GLOBAL_LOAD_ASYNC_TO_LDS_B128 (ASYNCcnt) path.
// ---------------------------------------------------------------------------

typedef __attribute__((ext_vector_type(16))) _Float16 v16h;
typedef __attribute__((ext_vector_type(8)))  _Float16 v8h;
typedef __attribute__((ext_vector_type(4)))  _Float16 v4h;
typedef __attribute__((ext_vector_type(8)))  float    v8f;
typedef __attribute__((ext_vector_type(4)))  int      v4i;

union HFrag { v16h v; v8h h[2]; };

__device__ __forceinline__ v8f wmma16(v16h a, v16h b, v8f c) {
  // v_wmma_f32_16x16x32_f16  D = A(16x32) * B(32x16) + C(16x16 f32)
  return __builtin_amdgcn_wmma_f32_16x16x32_f16(false, a, false, b,
                                                (short)0, c, false, false);
}

#define AS1 __attribute__((address_space(1)))
#define AS3 __attribute__((address_space(3)))

// 16-byte global -> LDS async copy (ASYNCcnt); fallback: plain copy.
__device__ __forceinline__ void async_b128(void* lds, const void* g) {
#if __has_builtin(__builtin_amdgcn_global_load_async_to_lds_b128)
  __builtin_amdgcn_global_load_async_to_lds_b128(
      (AS1 v4i*)(v4i*)(void*)g, (AS3 v4i*)(v4i*)lds, 0, 0);
#else
  *(v8h*)lds = *(const v8h*)g;
#endif
}

__device__ __forceinline__ void wait_async0() {
#if __has_builtin(__builtin_amdgcn_s_wait_asynccnt)
  __builtin_amdgcn_s_wait_asynccnt(0);
#elif __has_builtin(__builtin_amdgcn_global_load_async_to_lds_b128)
  asm volatile("s_wait_asynccnt 0" ::: "memory");
#endif
}

static constexpr int BB = 4;     // batch
static constexpr int NN = 2048;  // sequence
static constexpr int CC = 1024;  // channels
static constexpr int HH = 16;    // heads
static constexpr int DD = 64;    // head dim

// ---------------------------------------------------------------------------
// fp32 -> fp16 conversion (4-wide)
// ---------------------------------------------------------------------------
__global__ void cvt_f32_f16_v4(const float* __restrict__ s,
                               _Float16* __restrict__ d, int n4) {
  int i = blockIdx.x * blockDim.x + threadIdx.x;
  if (i < n4) {
    float4 v = ((const float4*)s)[i];
    v4h o = {(_Float16)v.x, (_Float16)v.y, (_Float16)v.z, (_Float16)v.w};
    ((v4h*)d)[i] = o;
  }
}

// ---------------------------------------------------------------------------
// Transposing fp32 -> fp16 convert: dst[n][k] = src[k][n]   (64x64 LDS tiles)
// ---------------------------------------------------------------------------
__global__ __launch_bounds__(256)
void transpose_f32_f16(const float* __restrict__ src, _Float16* __restrict__ dst,
                       int K, int N) {
  __shared__ _Float16 tile[64][64];
  const int n0 = blockIdx.x * 64, k0 = blockIdx.y * 64;
  const int tx = threadIdx.x & 63, ty = threadIdx.x >> 6;  // ty: 0..3
#pragma unroll
  for (int r = 0; r < 16; ++r) {
    int k = ty * 16 + r;
    tile[tx][k] = (_Float16)src[(size_t)(k0 + k) * N + n0 + tx];
  }
  __syncthreads();
#pragma unroll
  for (int r = 0; r < 16; ++r) {
    int n = ty * 16 + r;
    dst[(size_t)(n0 + n) * K + k0 + tx] = tile[n][tx];
  }
}

// ---------------------------------------------------------------------------
// GEMM core: C[M x Ncols] = A[M x 1024] * Wt[Ncols x 1024]^T
// Block tile 128x128, K-tile 32, double-buffered async LDS staging.
// 8 waves (2 x 4), wave tile 64x32 -> 8 WMMA per K-step.
// ---------------------------------------------------------------------------
template <typename EPI>
__device__ __forceinline__ void gemm128(const _Float16* __restrict__ Xp,
                                        const _Float16* __restrict__ Wtp,
                                        int m0, int n0, EPI&& epilogue) {
  __shared__ _Float16 As[2][128][32];   // [m][k]
  __shared__ _Float16 Bt[2][128][32];   // [n][k]

  const int tid  = threadIdx.x;
  const int lane = tid & 31;
  const int wave = tid >> 5;
  const int half = lane >> 4;
  const int l15  = lane & 15;
  const int wm = wave & 1;
  const int wn = wave >> 1;

  // one 16B chunk per thread per array: row = tid>>1, part = tid&1
  const int crow = tid >> 1, cpart = (tid & 1) * 16;
  const _Float16* ag = Xp  + (size_t)(m0 + crow) * CC + cpart;
  const _Float16* bg = Wtp + (size_t)(n0 + crow) * CC + cpart;

  v8f acc[4][2] = {};

  // prologue: stage tile 0 into buffer 0
  async_b128(&As[0][crow][cpart], ag);
  async_b128(&Bt[0][crow][cpart], bg);

  int buf = 0;
  for (int k0 = 0; k0 < CC; k0 += 32) {
    wait_async0();
    __syncthreads();
    if (k0 + 32 < CC) {
      async_b128(&As[buf ^ 1][crow][cpart], ag + k0 + 32);
      async_b128(&Bt[buf ^ 1][crow][cpart], bg + k0 + 32);
    }

    HFrag a[4], b[2];
#pragma unroll
    for (int ms = 0; ms < 4; ++ms) {
      int row = wm * 64 + ms * 16 + l15;
      a[ms].h[0] = *(const v8h*)&As[buf][row][half * 8];
      a[ms].h[1] = *(const v8h*)&As[buf][row][16 + half * 8];
    }
#pragma unroll
    for (int ns = 0; ns < 2; ++ns) {
      int col = wn * 32 + ns * 16 + l15;
      b[ns].h[0] = *(const v8h*)&Bt[buf][col][half * 8];
      b[ns].h[1] = *(const v8h*)&Bt[buf][col][16 + half * 8];
    }
#pragma unroll
    for (int ms = 0; ms < 4; ++ms)
#pragma unroll
      for (int ns = 0; ns < 2; ++ns)
        acc[ms][ns] = wmma16(a[ms].v, b[ns].v, acc[ms][ns]);
    buf ^= 1;
  }

#pragma unroll
  for (int ms = 0; ms < 4; ++ms)
#pragma unroll
    for (int ns = 0; ns < 2; ++ns)
#pragma unroll
      for (int r = 0; r < 8; ++r) {
        int token = m0 + wm * 64 + ms * 16 + (half ? r + 8 : r);
        int c     = n0 + wn * 32 + ns * 16 + l15;
        epilogue(token, c, acc[ms][ns][r]);
      }
}

// QKV projection: epilogue scatters Q,K [B,H,N,D] and V^T [B,H,D,N] as f16.
__global__ __launch_bounds__(256)
void qkv_gemm(const _Float16* __restrict__ X, const _Float16* __restrict__ Wt,
              const float* __restrict__ bias,
              _Float16* __restrict__ Qo, _Float16* __restrict__ Ko,
              _Float16* __restrict__ Vt) {
  const int m0 = blockIdx.y * 128;
  const int n0 = blockIdx.x * 128;
  gemm128(X, Wt, m0, n0, [&](int token, int c, float v) {
    _Float16 hv = (_Float16)(v + bias[c]);
    int t = c >> 10, r0 = c & 1023, h = r0 >> 6, d = r0 & 63;
    int bb = token >> 11, n = token & (NN - 1);
    size_t bh = (size_t)(bb * HH + h);
    if (t == 0)      Qo[(bh * NN + n) * DD + d] = hv;
    else if (t == 1) Ko[(bh * NN + n) * DD + d] = hv;
    else             Vt[(bh * DD + d) * NN + n] = hv;
  });
}

// Output projection: f32 output + bias.
__global__ __launch_bounds__(256)
void proj_gemm(const _Float16* __restrict__ A, const _Float16* __restrict__ Wt,
               const float* __restrict__ bias, float* __restrict__ out) {
  const int m0 = blockIdx.y * 128;
  const int n0 = blockIdx.x * 128;
  gemm128(A, Wt, m0, n0, [&](int token, int c, float v) {
    out[(size_t)token * CC + c] = v + bias[c];
  });
}

// ---------------------------------------------------------------------------
// Flash attention: one block per (b,h, 128-query tile).
// 8 waves x 16 query rows; 32-key tiles double-buffered via async LDS.
// ---------------------------------------------------------------------------
__global__ __launch_bounds__(256)
void attn_kernel(const _Float16* __restrict__ Q, const _Float16* __restrict__ K,
                 const _Float16* __restrict__ Vt,
                 const unsigned char* __restrict__ mask,
                 _Float16* __restrict__ Aout) {
  __shared__ _Float16 Ks[2][32][64];     // [key][d]
  __shared__ _Float16 Vts[2][64][32];    // [d][key]
  __shared__ _Float16 Pst[8][16][32];    // per-wave P re-layout bounce

  const int tid  = threadIdx.x;
  const int lane = tid & 31;
  const int wave = tid >> 5;
  const int half = lane >> 4;
  const int l15  = lane & 15;

  const int bh = blockIdx.x;             // 0..63
  const int bb = bh / HH;
  const int hh = bh % HH;
  const int q0 = blockIdx.y * 128;
  const float scale = 0.125f;            // D^-0.5

  const _Float16* Qp = Q  + (size_t)bh * NN * DD;
  const _Float16* Kp = K  + (size_t)bh * NN * DD;
  const _Float16* Vp = Vt + (size_t)bh * DD * NN;

  // staging chunk assignment (one 16B chunk per thread per array)
  const int kkey = tid >> 3, kpart = (tid & 7) * 8;   // Ks: 32 rows x 8 chunks
  const int vd   = tid >> 2, vpart = (tid & 3) * 8;   // Vts: 64 rows x 4 chunks
  const _Float16* kg = Kp + (size_t)kkey * DD + kpart;
  const _Float16* vg = Vp + (size_t)vd * NN + vpart;

  // Q fragments (16 rows x 64 dims -> two 16x32 A-frags), kept in registers
  HFrag qa[2];
  const int qrow = q0 + wave * 16 + l15;
#pragma unroll
  for (int c = 0; c < 2; ++c) {
    qa[c].h[0] = *(const v8h*)(Qp + (size_t)qrow * DD + c * 32 + half * 8);
    qa[c].h[1] = *(const v8h*)(Qp + (size_t)qrow * DD + c * 32 + 16 + half * 8);
  }

  v8f O[4] = {};
  float mrow[8], lrow[8];
#pragma unroll
  for (int r = 0; r < 8; ++r) { mrow[r] = -1e30f; lrow[r] = 0.0f; }

  // prologue: stage key-tile 0
  async_b128(&Ks[0][kkey][kpart], kg);
  async_b128(&Vts[0][vd][vpart], vg);

  int buf = 0;
  for (int k0 = 0; k0 < NN; k0 += 32) {
    wait_async0();
    __syncthreads();
    if (k0 + 32 < NN) {
      async_b128(&Ks[buf ^ 1][kkey][kpart], kg + (size_t)(k0 + 32) * DD);
      async_b128(&Vts[buf ^ 1][vd][vpart], vg + k0 + 32);
    }

    // S = Q * K^T  (two 16x16 key sub-tiles, contraction over D=64)
    v8f S[2] = {};
#pragma unroll
    for (int ns = 0; ns < 2; ++ns) {
      int key = ns * 16 + l15;
      HFrag kb;
#pragma unroll
      for (int c = 0; c < 2; ++c) {
        kb.h[0] = *(const v8h*)&Ks[buf][key][c * 32 + half * 8];
        kb.h[1] = *(const v8h*)&Ks[buf][key][c * 32 + 16 + half * 8];
        S[ns] = wmma16(qa[c].v, kb.v, S[ns]);
      }
    }

    const float mk0 = mask[bb * NN + k0 + l15]      ? 1.0f : 0.0f;
    const float mk1 = mask[bb * NN + k0 + 16 + l15] ? 1.0f : 0.0f;

    float alpha[8];
#pragma unroll
    for (int r = 0; r < 8; ++r) {
      float s0 = S[0][r] * scale, s1 = S[1][r] * scale;
      float t = fmaxf(mk0 ? s0 : -1e30f, mk1 ? s1 : -1e30f);
      t = fmaxf(t, __shfl_xor(t, 1));
      t = fmaxf(t, __shfl_xor(t, 2));
      t = fmaxf(t, __shfl_xor(t, 4));
      t = fmaxf(t, __shfl_xor(t, 8));
      float mnew = fmaxf(mrow[r], t);
      float a = __expf(mrow[r] - mnew);
      float p0 = mk0 ? __expf(s0 - mnew) : 0.0f;
      float p1 = mk1 ? __expf(s1 - mnew) : 0.0f;
      float rs = p0 + p1;
      rs += __shfl_xor(rs, 1);
      rs += __shfl_xor(rs, 2);
      rs += __shfl_xor(rs, 4);
      rs += __shfl_xor(rs, 8);
      lrow[r] = lrow[r] * a + rs;
      mrow[r] = mnew;
      alpha[r] = a;
      S[0][r] = p0; S[1][r] = p1;
    }
#pragma unroll
    for (int n = 0; n < 4; ++n)
#pragma unroll
      for (int r = 0; r < 8; ++r) O[n][r] *= alpha[r];

    // P: C-frag layout -> A-frag layout via per-wave LDS bounce
#pragma unroll
    for (int r = 0; r < 8; ++r) {
      int row = half ? r + 8 : r;
      Pst[wave][row][l15]      = (_Float16)S[0][r];
      Pst[wave][row][16 + l15] = (_Float16)S[1][r];
    }
    HFrag pa;                            // per-wave LDS is in-order
    pa.h[0] = *(const v8h*)&Pst[wave][l15][half * 8];
    pa.h[1] = *(const v8h*)&Pst[wave][l15][16 + half * 8];

    // O += P(16x32) * V(32x64)
#pragma unroll
    for (int n = 0; n < 4; ++n) {
      int dim = n * 16 + l15;
      HFrag vb;
      vb.h[0] = *(const v8h*)&Vts[buf][dim][half * 8];
      vb.h[1] = *(const v8h*)&Vts[buf][dim][16 + half * 8];
      O[n] = wmma16(pa.v, vb.v, O[n]);
    }
    buf ^= 1;
  }

  // normalize; fully-masked rows (l==0) -> 0 (matches nan_to_num)
  float inv[8];
#pragma unroll
  for (int r = 0; r < 8; ++r) inv[r] = lrow[r] > 0.0f ? 1.0f / lrow[r] : 0.0f;
#pragma unroll
  for (int n = 0; n < 4; ++n)
#pragma unroll
    for (int r = 0; r < 8; ++r) {
      int row = half ? r + 8 : r;
      int nq  = q0 + wave * 16 + row;
      int dim = n * 16 + l15;
      Aout[((size_t)(bb * NN + nq)) * CC + hh * DD + dim] =
          (_Float16)(O[n][r] * inv[r]);
    }
}

// ---------------------------------------------------------------------------
extern "C" void kernel_launch(void* const* d_in, const int* in_sizes, int n_in,
                              void* d_out, int out_size, void* d_ws,
                              size_t ws_size, hipStream_t stream) {
  const float*         x     = (const float*)d_in[0];
  const unsigned char* mask  = (const unsigned char*)d_in[1];  // jax bool = 1B
  const float*         w_qkv = (const float*)d_in[2];
  const float*         b_qkv = (const float*)d_in[3];
  const float*         w_out = (const float*)d_in[4];
  const float*         b_out = (const float*)d_in[5];
  float*               out   = (float*)d_out;

  char* ws = (char*)d_ws;
  size_t off = 0;
  auto alloc = [&](size_t bytes) {
    char* p = ws + off;
    off += (bytes + 255) & ~(size_t)255;
    return p;
  };
  const size_t nTok = (size_t)BB * NN;                        // 8192
  _Float16* xh     = (_Float16*)alloc(nTok * CC * 2);         // [tok][k]
  _Float16* wqkvt  = (_Float16*)alloc((size_t)3 * CC * CC * 2); // [n][k]
  _Float16* woutt  = (_Float16*)alloc((size_t)CC * CC * 2);     // [n][k]
  _Float16* Qh     = (_Float16*)alloc(nTok * CC * 2);         // [B,H,N,D]
  _Float16* Kh     = (_Float16*)alloc(nTok * CC * 2);         // [B,H,N,D]
  _Float16* Vth    = (_Float16*)alloc(nTok * CC * 2);         // [B,H,D,N]
  _Float16* Ah     = (_Float16*)alloc(nTok * CC * 2);         // [B,N,C]

  // x: fp32 -> f16; weights: fp32 -> f16 transposed to [n][k]
  {
    int n4 = (int)(nTok * CC / 4);
    cvt_f32_f16_v4<<<(n4 + 255) / 256, 256, 0, stream>>>(x, xh, n4);
    transpose_f32_f16<<<dim3(3 * CC / 64, CC / 64), 256, 0, stream>>>(
        w_qkv, wqkvt, CC, 3 * CC);
    transpose_f32_f16<<<dim3(CC / 64, CC / 64), 256, 0, stream>>>(
        w_out, woutt, CC, CC);
  }

  // QKV projection
  qkv_gemm<<<dim3(3 * CC / 128, nTok / 128), 256, 0, stream>>>(
      xh, wqkvt, b_qkv, Qh, Kh, Vth);

  // Flash attention
  attn_kernel<<<dim3(BB * HH, NN / 128), 256, 0, stream>>>(
      Qh, Kh, Vth, mask, Ah);

  // Output projection
  proj_gemm<<<dim3(CC / 128, nTok / 128), 256, 0, stream>>>(
      Ah, woutt, b_out, out);
}
